// TernaryLinear_74612171866703
// MI455X (gfx1250) — matrix operand order repaired and checked by reading, not compiled, not run on hardware
//
#include <hip/hip_runtime.h>
#include <hip/hip_bf16.h>

// ---------------------------------------------------------------------------
// TernaryLinear for MI455X (gfx1250, wave32, WMMA + async-LDS DMA).
//   gamma = mean(|W|);  Wq = clip(round(W/gamma),-1,1) stored as bf16
//   Y = (Xbf16 @ Wq^T) * gamma + bias      via V_WMMA_F32_16X16X32_BF16
// M=B*S=16384, N=D_OUT=16384, K=D_IN=4096.
// Roofline: 2.2 TFLOP vs ~1.6 GB min HBM traffic -> memory bound at 23.3 TB/s.
// Strategy: bf16 ternary weights (128 MB) stay L2-resident; X tiles shared
// across a grid row via L2 (blockIdx.x fastest). Double-buffered LDS pipeline:
//   W tile : global_load_async_to_lds_b128 (ASYNCcnt DMA, zero VALU decode)
//   X tile : register-staged + v_cvt_pk_bf16_f32 (needs fp32->bf16 convert)
// global_prefetch_b8 runs ~8 stages ahead on both streams.
// ---------------------------------------------------------------------------

typedef __attribute__((ext_vector_type(16))) __bf16 v16bf;
typedef __attribute__((ext_vector_type(8)))  __bf16 v8bf;
typedef __attribute__((ext_vector_type(4)))  __bf16 v4bf;
typedef __attribute__((ext_vector_type(8)))  float  v8f;
typedef __attribute__((ext_vector_type(4)))  float  v4f;

#define K_TOT 4096
#define N_TOT 16384
#define M_TOT 16384
#define W_ELEMS (N_TOT * K_TOT)      // 67108864 weights
#define RED_BLOCKS 1024

#define BM 128
#define BN 128
#define BK 32
#define NSTEP (K_TOT / BK)           // 128 pipeline stages

#define LDS_AS __attribute__((address_space(3)))

// global -> LDS async DMA, 16 bytes/lane; INST_OFFSET applies to BOTH the LDS
// destination and the global source address (ISA 10.x / 15.18.3), so two
// instructions with offset 0/16 share the same address registers.
__device__ __forceinline__ void async_w_tile(unsigned lds_addr, const void* gaddr) {
  unsigned long long ga = (unsigned long long)(uintptr_t)gaddr;
  asm volatile("global_load_async_to_lds_b128 %0, %1, off\n\t"
               "global_load_async_to_lds_b128 %0, %1, off offset:16"
               :: "v"(lds_addr), "v"(ga) : "memory");
}
__device__ __forceinline__ void wait_async0() {
  asm volatile("s_wait_asynccnt 0x0" ::: "memory");
}

// ---------------------------------------------------------------- reductions
__global__ __launch_bounds__(256) void reduce_abs_kernel(
    const float* __restrict__ w, float* __restrict__ partial) {
  __shared__ float sm[256];
  float s = 0.0f;
  int t = blockIdx.x * 256 + threadIdx.x;
  const int stride = RED_BLOCKS * 256;
  #pragma unroll 4
  for (int i = 0; i < W_ELEMS / stride; ++i)
    s += fabsf(w[t + i * stride]);
  sm[threadIdx.x] = s;
  __syncthreads();
  for (int o = 128; o > 0; o >>= 1) {
    if (threadIdx.x < o) sm[threadIdx.x] += sm[threadIdx.x + o];
    __syncthreads();
  }
  if (threadIdx.x == 0) partial[blockIdx.x] = sm[0];
}

__global__ __launch_bounds__(256) void finalize_gamma_kernel(
    const float* __restrict__ partial, float* __restrict__ gamma) {
  __shared__ float sm[256];
  int t = threadIdx.x;
  float s = partial[t] + partial[t + 256] + partial[t + 512] + partial[t + 768];
  sm[t] = s;
  __syncthreads();
  for (int o = 128; o > 0; o >>= 1) {
    if (t < o) sm[t] += sm[t + o];
    __syncthreads();
  }
  if (t == 0) gamma[0] = fmaxf(sm[0] / (float)W_ELEMS, 1e-9f);
}

// ---------------------------------------------------------------- quantize
// Wq[i] = clamp(round(W[i]/gamma), -1, 1) stored as bf16 (exact: {-1,0,1});
// gamma is applied once in the GEMM epilogue.
__global__ __launch_bounds__(256) void quantize_w_kernel(
    const float* __restrict__ w, const float* __restrict__ gammap,
    __bf16* __restrict__ wq) {
  const float inv = 1.0f / gammap[0];
  const int i = (blockIdx.x * 256 + threadIdx.x) * 4;
  v4f v = *(const v4f*)(w + i);
  v4bf q;
  #pragma unroll
  for (int j = 0; j < 4; ++j) {
    float t = rintf(v[j] * inv);
    q[j] = (__bf16)fminf(1.0f, fmaxf(-1.0f, t));
  }
  *(v4bf*)(wq + i) = q;   // 8-byte store
}

// ---------------------------------------------------------------- GEMM
// Block: 128x128 tile, 256 threads = 8 wave32 waves in 4(M) x 2(N) grid.
// Each wave: 32x64 sub-tile = 2x4 WMMA 16x16 f32 accumulators.
__global__ __launch_bounds__(256) void gemm_wmma_bf16_kernel(
    const float* __restrict__ x, const __bf16* __restrict__ wq,
    const float* __restrict__ bias, const float* __restrict__ gammap,
    float* __restrict__ out) {
  __shared__ __align__(16) __bf16 Xs[2][BM * BK];   // 2 x 8 KB, [row m][k]
  __shared__ __align__(16) __bf16 Ws[2][BN * BK];   // 2 x 8 KB, [row n][k]

  const int tid  = threadIdx.x;
  const int lane = tid & 31;
  const int wave = tid >> 5;
  const int waveM = wave & 3;   // 0..3 -> 32-row slab
  const int waveN = wave >> 2;  // 0..1 -> 64-col slab
  const int mTile = blockIdx.y * BM;
  const int nTile = blockIdx.x * BN;

  // cooperative-load coordinates: each thread moves 16 K-elements of one row
  const int lr = tid >> 1;          // 0..127
  const int lc = (tid & 1) * 16;    // 0 or 16

  const float*  gx = x  + (size_t)(mTile + lr) * K_TOT + lc;
  const __bf16* gw = wq + (size_t)(nTile + lr) * K_TOT + lc;

  // 32-bit LDS byte addresses for this thread's W-tile slice, per buffer
  const unsigned wlds0 =
      (unsigned)(uintptr_t)(LDS_AS __bf16*)&Ws[0][lr * BK + lc];
  const unsigned wlds1 =
      (unsigned)(uintptr_t)(LDS_AS __bf16*)&Ws[1][lr * BK + lc];

  v8f acc[2][4];
  #pragma unroll
  for (int i = 0; i < 2; ++i)
    #pragma unroll
    for (int j = 0; j < 4; ++j) acc[i][j] = v8f{};

  // ISA 7.12.2 fragment gathers (wave32):
  // A 16x32 bf16: lanes 0-15 row=lane, K {0..7,16..23}; lanes 16-31 row=lane-16,
  //               K {8..15,24..31}  -> ka0 = (lane>>4)*8, halves at ka0, ka0+16.
  // B 32x16 bf16: lanes 0-15 col=lane, K {0..7,16..23}; lanes 16-31 K {8..15,24..31}
  //               -> kb0 = (lane>>4)*8, halves at kb0, kb0+16.
  const int rowSel = lane & 15;
  const int ka0 = (lane >> 4) * 8;
  const int kb0 = (lane >> 4) * 8;

  // register staging for the X pipeline
  v4f xf0, xf1, xf2, xf3;

  auto store_x_tile = [&](int buf) {
    v8bf a0, a1;
    #pragma unroll
    for (int j = 0; j < 4; ++j) {
      a0[j]     = (__bf16)xf0[j];
      a0[4 + j] = (__bf16)xf1[j];
      a1[j]     = (__bf16)xf2[j];
      a1[4 + j] = (__bf16)xf3[j];
    }
    *(v8bf*)(&Xs[buf][lr * BK + lc])     = a0;
    *(v8bf*)(&Xs[buf][lr * BK + lc + 8]) = a1;
  };

  // ---- prologue: stage 0 into buffer 0
  async_w_tile(wlds0, gw);
  xf0 = *(const v4f*)(gx + 0);
  xf1 = *(const v4f*)(gx + 4);
  xf2 = *(const v4f*)(gx + 8);
  xf3 = *(const v4f*)(gx + 12);
  store_x_tile(0);
  wait_async0();
  __syncthreads();

  // ---- main pipeline: load(k+1) || compute(k)
  for (int kb = 0; kb < NSTEP; ++kb) {
    const int cur = kb & 1;

    if (kb + 1 < NSTEP) {
      // W tile kb+1: async DMA straight into the other LDS buffer
      async_w_tile(cur ? wlds0 : wlds1, gw + (size_t)(kb + 1) * BK);
      // X tile kb+1: global -> registers (converted after compute)
      const float* nx = gx + (size_t)(kb + 1) * BK;
      xf0 = *(const v4f*)(nx + 0);
      xf1 = *(const v4f*)(nx + 4);
      xf2 = *(const v4f*)(nx + 8);
      xf3 = *(const v4f*)(nx + 12);
      // speculative prefetch ~8 stages ahead (global_prefetch_b8; OOB dropped)
      __builtin_prefetch(gx + (size_t)(kb + 9) * BK, 0, 1);
      __builtin_prefetch(gw + (size_t)(kb + 9) * BK, 0, 1);
    }

    // compute on buffer `cur`
    v16bf afr[2], bfr[4];
    #pragma unroll
    for (int mi = 0; mi < 2; ++mi) {
      const int row = waveM * 32 + mi * 16 + rowSel;
      v8bf lo = *(const v8bf*)(&Xs[cur][row * BK + ka0]);
      v8bf hi = *(const v8bf*)(&Xs[cur][row * BK + ka0 + 16]);
      afr[mi] = __builtin_shufflevector(lo, hi, 0, 1, 2, 3, 4, 5, 6, 7,
                                        8, 9, 10, 11, 12, 13, 14, 15);
    }
    #pragma unroll
    for (int ni = 0; ni < 4; ++ni) {
      const int row = waveN * 64 + ni * 16 + rowSel;
      v8bf lo = *(const v8bf*)(&Ws[cur][row * BK + kb0]);
      v8bf hi = *(const v8bf*)(&Ws[cur][row * BK + kb0 + 16]);
      bfr[ni] = __builtin_shufflevector(lo, hi, 0, 1, 2, 3, 4, 5, 6, 7,
                                        8, 9, 10, 11, 12, 13, 14, 15);
    }
    #pragma unroll
    for (int mi = 0; mi < 2; ++mi)
      #pragma unroll
      for (int ni = 0; ni < 4; ++ni)
        acc[mi][ni] = __builtin_amdgcn_wmma_f32_16x16x32_bf16(
            /*neg_a=*/false, afr[mi], /*neg_b=*/false, bfr[ni],
            /*c_mod=*/(short)0, acc[mi][ni],
            /*reuse_a=*/false, /*reuse_b=*/false);

    // publish stage kb+1; one barrier per iteration
    if (kb + 1 < NSTEP) store_x_tile(cur ^ 1);
    wait_async0();
    __syncthreads();
  }

  // Epilogue: C/D f32 layout (VGPR r: lanes 0-15 M=r, lanes 16-31 M=8+r; N=lane&15)
  const float g = gammap[0];
  #pragma unroll
  for (int mi = 0; mi < 2; ++mi) {
    const int mbase = mTile + waveM * 32 + mi * 16 + (lane >> 4) * 8;
    #pragma unroll
    for (int ni = 0; ni < 4; ++ni) {
      const int n = nTile + waveN * 64 + ni * 16 + (lane & 15);
      const float bz = bias[n];
      #pragma unroll
      for (int r = 0; r < 8; ++r)
        out[(size_t)(mbase + r) * N_TOT + n] = acc[mi][ni][r] * g + bz;
    }
  }
}

// ---------------------------------------------------------------- launch
extern "C" void kernel_launch(void* const* d_in, const int* in_sizes, int n_in,
                              void* d_out, int out_size, void* d_ws, size_t ws_size,
                              hipStream_t stream) {
  const float* x    = (const float*)d_in[0];   // [16384, 4096]
  const float* w    = (const float*)d_in[1];   // [16384, 4096]
  const float* bias = (const float*)d_in[2];   // [16384]
  float* out        = (float*)d_out;           // [16384, 16384]

  // ws layout: [0, 128MB) bf16 Wq | 1024 float partials | 1 float gamma
  __bf16* wq     = (__bf16*)d_ws;
  float* partial = (float*)((char*)d_ws + (size_t)W_ELEMS * sizeof(__bf16));
  float* gamma   = partial + RED_BLOCKS;

  reduce_abs_kernel<<<RED_BLOCKS, 256, 0, stream>>>(w, partial);
  finalize_gamma_kernel<<<1, 256, 0, stream>>>(partial, gamma);
  quantize_w_kernel<<<W_ELEMS / 1024, 256, 0, stream>>>(w, gamma, wq);

  dim3 grid(N_TOT / BN, M_TOT / BM);   // x fastest -> grid rows share X tile in L2
  gemm_wmma_bf16_kernel<<<grid, 256, 0, stream>>>(x, wq, bias, gamma, out);
}
